// TimeSformer_26104811225128
// MI455X (gfx1250) — compile-verified, matrix operand order checked by reading, not run on hardware
//
#include <hip/hip_runtime.h>
#include <hip/hip_bf16.h>
#include <math.h>
#include <stdint.h>

// ---------------- model constants ----------------
#define D_    768
#define NH_   12
#define HD_   64
#define T_    8
#define B_    2
#define NTOK  197           // 196 patches + cls
#define NPAD  208           // padded to 13 WMMA tiles
#define KPAD  224           // padded K for P@V (7 x 32)
#define MP    (B_*T_*NPAD)  // 3328 padded tokens total
#define MT    (MP/16)       // 208 M tiles (divisible by 4)
#define MLPH_ 3072
#define NPATCH 196
#define MIM   (B_*T_*NPATCH) // 3136 im2col rows (196 M tiles, divisible by 4)
#define KC    128            // K chunk staged per async double-buffer step

typedef __attribute__((ext_vector_type(16))) __bf16 v16bf;
typedef __attribute__((ext_vector_type(8)))  float  v8f;

union Frag { v16bf v; uint4 q[2]; };

__device__ __forceinline__ unsigned short f2bf(float f){
  unsigned int u = __float_as_uint(f);
  u += 0x7FFFu + ((u >> 16) & 1u);        // round-to-nearest-even
  return (unsigned short)(u >> 16);
}

// Load one 16x32 bf16 A/B fragment row for this lane:
// elements [k0+8*half .. +7] and [k0+16+8*half .. +7]   (per CDNA5 ISA layout)
__device__ __forceinline__ void ldfrag(Frag& f, const unsigned short* rowp, int k0, int half){
  f.q[0] = *(const uint4*)(rowp + k0 + 8*half);
  f.q[1] = *(const uint4*)(rowp + k0 + 16 + 8*half);
}

// Async-stage one 16 x KC bf16 weight chunk into LDS (ASYNCcnt tracked).
// 256 x b128 transfers, 2 per thread for a 128-thread block.
__device__ __forceinline__ void stage_async(const unsigned short* W, int ldw, int n0, int kc,
                                            unsigned short* buf, int tid){
#pragma unroll
  for (int t = tid; t < 256; t += 128){
    const int row = t >> 4;            // 0..15
    const int col = (t & 15) * 8;      // 0..120
    const unsigned short* gp = W + (size_t)(n0 + row)*ldw + kc + col;
    const unsigned ldsa = (unsigned)(uintptr_t)(buf + row*KC + col);
    asm volatile("global_load_async_to_lds_b128 %0, %1, off"
                 :: "v"(ldsa), "v"(gp) : "memory");
  }
}
__device__ __forceinline__ void wait_async(){
  asm volatile("s_wait_asynccnt 0x0" ::: "memory");
}

// ---------------- generic WMMA GEMM:  C[M,N] = A[M,K](bf16) x W[N,K](bf16)^T ----------------
// 4 waves / block, all sharing one 16-row weight tile staged via async double-buffer.
struct GArgs {
  const unsigned short* A; int lda;
  const unsigned short* W;            // [Nout, K] bf16 (pre-transposed)
  const float* bias;
  int Mtiles, Ntiles, K;              // Mtiles % 4 == 0, K % KC == 0
  float* H;                           // residual target / patch dest
  unsigned short* outbf; int ldo;     // bf16 dest (gelu)
  unsigned short *q, *k, *vT;         // qkv scatter dests
  const float* pos;                   // pos_embed (patch epilogue)
};

enum { EP_PATCH = 0, EP_QKV = 1, EP_RES = 2, EP_GELU = 3 };

template<int EP>
__global__ __launch_bounds__(128) void k_gemm(GArgs g){
  __shared__ __align__(16) unsigned short wbuf[2][16*KC];   // 8 KB double buffer
  const int tid  = threadIdx.x;
  const int wave = tid >> 5;
  const int lane = tid & 31;
  const int half = lane >> 4;
  const int r    = lane & 15;
  const int mblk = g.Mtiles >> 2;
  const int nt   = blockIdx.x / mblk;
  const int mt   = (blockIdx.x - nt*mblk)*4 + wave;
  const int m0   = mt*16, n0 = nt*16;
  const unsigned short* Ar = g.A + (size_t)(m0 + r) * g.lda;

  stage_async(g.W, g.K, n0, 0, &wbuf[0][0], tid);           // prologue: chunk 0 in flight
  v8f acc = {};
  int ib = 0;
  for (int kc = 0; kc < g.K; kc += KC, ib ^= 1){
    wait_async();                                           // own async issues done
    __syncthreads();                                        // all waves' chunk visible
    if (kc + KC < g.K)                                      // overlap: next chunk in flight
      stage_async(g.W, g.K, n0, kc + KC, &wbuf[ib^1][0], tid);
    __builtin_prefetch(Ar + kc + KC, 0, 0);                 // global_prefetch on A stream
    const unsigned short* Bb = &wbuf[ib][r*KC];
#pragma unroll
    for (int k0 = 0; k0 < KC; k0 += 32){
      Frag a, b;
      ldfrag(a, Ar, kc + k0, half);                         // global_load_b128
      ldfrag(b, Bb, k0, half);                              // ds_load_b128
      acc = __builtin_amdgcn_wmma_f32_16x16x32_bf16(false, a.v, false, b.v,
                                                    (short)0, acc, false, false);
    }
  }
#pragma unroll
  for (int j = 0; j < 8; j++){
    const int m = m0 + j + 8*half;              // C layout: VGPR j -> M = j + 8*half
    const int n = n0 + r;                       // col = lane&15
    float v = acc[j];
    if (EP == EP_PATCH){
      const int bt = m / NPATCH, np = m - bt*NPATCH, t = bt & 7;
      v += g.bias[n] + g.pos[((size_t)(t*NTOK + 1 + np))*D_ + n];
      g.H[((size_t)bt*NPAD + 1 + np)*D_ + n] = v;
    } else if (EP == EP_QKV){
      v += g.bias[n];
      const int token = m % NPAD, bt = m / NPAD;
      const int which = n / D_, rem = n - which*D_;
      const int h = rem >> 6, d = rem & 63;
      const size_t hb = (size_t)bt*NH_ + h;
      const unsigned short bv = f2bf(v);
      if (which == 0)      g.q [(hb*NPAD + token)*HD_ + d] = bv;
      else if (which == 1) g.k [(hb*NPAD + token)*HD_ + d] = bv;
      else                 g.vT[(hb*HD_  + d)*KPAD + token] = bv;   // transposed V
    } else if (EP == EP_RES){
      v += g.bias[n];
      const size_t idx = (size_t)m*D_ + n;
      g.H[idx] += v;                            // residual add
    } else {                                    // EP_GELU (exact erf gelu)
      v += g.bias[n];
      const float gl = 0.5f * v * (1.f + erff(v * 0.70710678118f));
      g.outbf[(size_t)m * g.ldo + n] = f2bf(gl);
    }
  }
}

// ---------------- fused attention: scores -> softmax -> P@V ----------------
__global__ __launch_bounds__(128) void k_attn(const unsigned short* q, const unsigned short* kk,
                                              const unsigned short* vT, unsigned short* out){
  __shared__ __align__(16) float          sS[16][KPAD];
  __shared__ __align__(16) unsigned short sP[16][KPAD];
  const int wave = threadIdx.x >> 5;
  const int lane = threadIdx.x & 31;
  const int half = lane >> 4;
  const int r    = lane & 15;
  const int strip = blockIdx.x % 13;
  const int bh    = blockIdx.x / 13;
  const int h  = bh % NH_;
  const int bt = bh / NH_;
  const size_t ho = (size_t)bt*NH_ + h;
  const unsigned short* qh = q  + ho*NPAD*HD_;
  const unsigned short* kh = kk + ho*NPAD*HD_;
  const unsigned short* vh = vT + ho*HD_*KPAD;
  const int m0 = strip * 16;

  // phase 1: S = q @ k^T * 1/8 for this 16-row strip, all 13 column tiles
  Frag a0, a1;
  ldfrag(a0, qh + (size_t)(m0 + r)*HD_, 0,  half);
  ldfrag(a1, qh + (size_t)(m0 + r)*HD_, 32, half);
  for (int j = wave; j < 13; j += 4){
    Frag b0, b1;
    const unsigned short* kr = kh + (size_t)(j*16 + r)*HD_;
    ldfrag(b0, kr, 0,  half);
    ldfrag(b1, kr, 32, half);
    v8f acc = {};
    acc = __builtin_amdgcn_wmma_f32_16x16x32_bf16(false, a0.v, false, b0.v, (short)0, acc, false, false);
    acc = __builtin_amdgcn_wmma_f32_16x16x32_bf16(false, a1.v, false, b1.v, (short)0, acc, false, false);
#pragma unroll
    for (int jj = 0; jj < 8; jj++)
      sS[jj + 8*half][j*16 + r] = acc[jj] * 0.125f;   // HD^-0.5
  }
  __syncthreads();

  // phase 2: masked softmax per row (mask columns >= 197)
  if (threadIdx.x < 16){
    const int row = threadIdx.x;
    float mx = -1e30f;
    for (int c = 0; c < NTOK; c++) mx = fmaxf(mx, sS[row][c]);
    float s = 0.f;
    for (int c = 0; c < NTOK; c++){ float e = __expf(sS[row][c] - mx); sS[row][c] = e; s += e; }
    const float inv = 1.f / s;
    for (int c = 0;    c < NTOK; c++) sP[row][c] = f2bf(sS[row][c] * inv);
    for (int c = NTOK; c < KPAD; c++) sP[row][c] = 0;
  }
  __syncthreads();

  // phase 3: out = P @ V  (wave w computes d-tile w of 4)
  {
    const int n0 = wave * 16;
    v8f acc = {};
    for (int k0 = 0; k0 < KPAD; k0 += 32){
      Frag a, b;
      ldfrag(a, &sP[r][0], k0, half);                       // ds_load_b128 fragments
      ldfrag(b, vh + (size_t)(n0 + r)*KPAD, k0, half);
      acc = __builtin_amdgcn_wmma_f32_16x16x32_bf16(false, a.v, false, b.v, (short)0, acc, false, false);
    }
#pragma unroll
    for (int jj = 0; jj < 8; jj++){
      const int token = m0 + jj + 8*half;
      out[((size_t)bt*NPAD + token)*D_ + h*HD_ + n0 + r] = f2bf(acc[jj]);
    }
  }
}

// ---------------- layernorm: f32 rows -> bf16 activation matrix ----------------
__global__ __launch_bounds__(256) void k_ln(const float* X, const float* w, const float* b,
                                            unsigned short* out){
  __shared__ float sm[8], sv[8], st[2];
  const int row = blockIdx.x;
  const float* xr = X + (size_t)row * D_;
  float s = 0.f, s2 = 0.f;
  for (int i = threadIdx.x; i < D_; i += 256){ float v = xr[i]; s += v; s2 += v*v; }
  for (int o = 16; o > 0; o >>= 1){ s += __shfl_down(s, o); s2 += __shfl_down(s2, o); }
  if ((threadIdx.x & 31) == 0){ sm[threadIdx.x >> 5] = s; sv[threadIdx.x >> 5] = s2; }
  __syncthreads();
  if (threadIdx.x == 0){
    float a = 0.f, c = 0.f;
    for (int i = 0; i < 8; i++){ a += sm[i]; c += sv[i]; }
    const float m = a / (float)D_;
    st[0] = m; st[1] = rsqrtf(c / (float)D_ - m*m + 1e-5f);
  }
  __syncthreads();
  const float m = st[0], iv = st[1];
  for (int i = threadIdx.x; i < D_; i += 256)
    out[(size_t)row*D_ + i] = f2bf((xr[i] - m) * iv * w[i] + b[i]);
}

// ---------------- small helpers ----------------
__global__ void k_convert(const float* in, unsigned short* out, int n){
  int i = blockIdx.x*256 + threadIdx.x;
  if (i < n) out[i] = f2bf(in[i]);
}
// transpose-convert: Wt[o*K+i] = bf16(W[i*N+o])   (W is [K,N] in-major)
__global__ void k_wt(const float* W, unsigned short* Wt, int K, int N){
  int idx = blockIdx.x*256 + threadIdx.x;
  if (idx >= N*K) return;
  int o = idx / K, i = idx - o*K;
  Wt[idx] = f2bf(W[(size_t)i*N + o]);
}
// im2col: A[m*768+k], m = bt*196+n, k = c*256+py*16+px
__global__ void k_im2col(const float* x, unsigned short* A){
  int idx = blockIdx.x*256 + threadIdx.x;
  if (idx >= MIM*D_) return;
  int m = idx / D_, kf = idx - m*D_;
  int c = kf >> 8, py = (kf >> 4) & 15, px = kf & 15;
  int bt = m / NPATCH, n = m - bt*NPATCH, b = bt >> 3, t = bt & 7;
  int gy = n / 14, gx = n - gy*14;
  A[idx] = f2bf(x[(((size_t)(b*3 + c)*T_ + t)*224 + gy*16 + py)*224 + gx*16 + px]);
}
// cls row + zero padded rows
__global__ void k_init(const float* cls, const float* pos, float* H){
  int idx = blockIdx.x*256 + threadIdx.x;
  if (idx >= B_*T_*12*D_) return;
  int d = idx % D_, rr = (idx / D_) % 12, bt = idx / (D_*12), t = bt & 7;
  if (rr == 0) H[((size_t)bt*NPAD)*D_ + d] = cls[d] + pos[((size_t)t*NTOK)*D_ + d];
  else         H[((size_t)bt*NPAD + 196 + rr)*D_ + d] = 0.f;
}
// final LN(row0) -> mean over frames -> head
__global__ __launch_bounds__(256) void k_head(const float* H, const float* nw, const float* nb,
                                              const float* hw, const float* hb, float* out){
  __shared__ float mean[16], inv[16];
  __shared__ float pooled[2][D_];
  const int tid = threadIdx.x;
  if (tid < 16){
    const float* row = H + (size_t)tid*NPAD*D_;
    float s = 0.f, s2 = 0.f;
    for (int i = 0; i < D_; i++){ float v = row[i]; s += v; s2 += v*v; }
    float m = s / (float)D_;
    mean[tid] = m; inv[tid] = rsqrtf(s2 / (float)D_ - m*m + 1e-5f);
  }
  __syncthreads();
  for (int idx = tid; idx < 2*D_; idx += 256){
    int b = idx / D_, d = idx - b*D_;
    float acc = 0.f;
    for (int t = 0; t < T_; t++){
      int bt = b*T_ + t;
      float v = H[((size_t)bt*NPAD)*D_ + d];
      acc += (v - mean[bt]) * inv[bt] * nw[d] + nb[d];
    }
    pooled[b][d] = acc * (1.f / (float)T_);
  }
  __syncthreads();
  for (int idx = tid; idx < 2*27; idx += 256){
    int b = idx / 27, c = idx - b*27;
    float acc = hb[c];
    for (int d = 0; d < D_; d++) acc += pooled[b][d] * hw[d*27 + c];
    out[b*27 + c] = acc;
  }
}

// ---------------- host ----------------
static inline int cdiv(long long a, long long b){ return (int)((a + b - 1) / b); }

extern "C" void kernel_launch(void* const* d_in, const int* in_sizes, int n_in,
                              void* d_out, int out_size, void* d_ws, size_t ws_size,
                              hipStream_t stream){
  (void)in_sizes; (void)n_in; (void)out_size; (void)ws_size;
  // input order (dict insertion order): x, conv_w, conv_b, cls_token, pos_embed,
  // 12 x {n1_w,n1_b,qkv_t_w,qkv_t_b,qkv_s_w,qkv_s_b,proj_w,proj_b,n2_w,n2_b,fc1_w,fc1_b,fc2_w,fc2_b},
  // norm_w, norm_b, head_w, head_b
  const float* x      = (const float*)d_in[0];
  const float* conv_w = (const float*)d_in[1];
  const float* conv_b = (const float*)d_in[2];
  const float* cls    = (const float*)d_in[3];
  const float* pos    = (const float*)d_in[4];
  const float* norm_w = (const float*)d_in[5 + 12*14 + 0];
  const float* norm_b = (const float*)d_in[5 + 12*14 + 1];
  const float* head_w = (const float*)d_in[5 + 12*14 + 2];
  const float* head_b = (const float*)d_in[5 + 12*14 + 3];

  char* ws = (char*)d_ws;
  size_t off = 0;
  auto alloc = [&](size_t bytes)->void*{
    void* p = ws + off; off = (off + bytes + 255) & ~(size_t)255; return p;
  };
  float*          H      = (float*)         alloc((size_t)MP*D_*4);
  unsigned short* A0     = (unsigned short*)alloc((size_t)MP*D_*2);
  unsigned short* A1     = (unsigned short*)alloc((size_t)MP*D_*2);
  unsigned short* Amlp   = (unsigned short*)alloc((size_t)MP*MLPH_*2);
  unsigned short* Wst    = (unsigned short*)alloc((size_t)MLPH_*D_*2);
  unsigned short* qB     = (unsigned short*)alloc((size_t)B_*T_*NH_*NPAD*HD_*2);
  unsigned short* kB     = (unsigned short*)alloc((size_t)B_*T_*NH_*NPAD*HD_*2);
  unsigned short* vTB    = (unsigned short*)alloc((size_t)B_*T_*NH_*HD_*KPAD*2);
  unsigned short* convbf = (unsigned short*)alloc((size_t)D_*D_*2);
  unsigned short* im2c   = (unsigned short*)alloc((size_t)MIM*D_*2);

  // ---- patch embed ----
  k_im2col<<<cdiv((long long)MIM*D_, 256), 256, 0, stream>>>(x, im2c);
  k_convert<<<cdiv((long long)D_*D_, 256), 256, 0, stream>>>(conv_w, convbf, D_*D_);
  {
    GArgs g{}; g.A = im2c; g.lda = D_; g.W = convbf; g.bias = conv_b;
    g.Mtiles = MIM/16; g.Ntiles = D_/16; g.K = D_; g.H = H; g.pos = pos;
    k_gemm<EP_PATCH><<<(g.Mtiles/4)*g.Ntiles, 128, 0, stream>>>(g);
  }
  k_init<<<cdiv((long long)B_*T_*12*D_, 256), 256, 0, stream>>>(cls, pos, H);

  // ---- transformer blocks ----
  for (int blk = 0; blk < 12; blk++){
    const float** bp = (const float**)(d_in + 5 + blk*14);
    const float* n1_w = bp[0],  *n1_b = bp[1];
    const float* qt_w = bp[2],  *qt_b = bp[3];
    const float* qs_w = bp[4],  *qs_b = bp[5];
    const float* pj_w = bp[6],  *pj_b = bp[7];
    const float* n2_w = bp[8],  *n2_b = bp[9];
    const float* f1_w = bp[10], *f1_b = bp[11];
    const float* f2_w = bp[12], *f2_b = bp[13];

    k_ln<<<MP, 256, 0, stream>>>(H, n1_w, n1_b, A0);

    // temporal MHA
    k_wt<<<cdiv((long long)3*D_*D_, 256), 256, 0, stream>>>(qt_w, Wst, D_, 3*D_);
    { GArgs g{}; g.A = A0; g.lda = D_; g.W = Wst; g.bias = qt_b;
      g.Mtiles = MT; g.Ntiles = 3*D_/16; g.K = D_; g.q = qB; g.k = kB; g.vT = vTB;
      k_gemm<EP_QKV><<<(g.Mtiles/4)*g.Ntiles, 128, 0, stream>>>(g); }
    k_attn<<<B_*T_*NH_*13, 128, 0, stream>>>(qB, kB, vTB, A1);

    // spatial MHA
    k_wt<<<cdiv((long long)3*D_*D_, 256), 256, 0, stream>>>(qs_w, Wst, D_, 3*D_);
    { GArgs g{}; g.A = A1; g.lda = D_; g.W = Wst; g.bias = qs_b;
      g.Mtiles = MT; g.Ntiles = 3*D_/16; g.K = D_; g.q = qB; g.k = kB; g.vT = vTB;
      k_gemm<EP_QKV><<<(g.Mtiles/4)*g.Ntiles, 128, 0, stream>>>(g); }
    k_attn<<<B_*T_*NH_*13, 128, 0, stream>>>(qB, kB, vTB, A0);

    // proj + residual
    k_wt<<<cdiv((long long)D_*D_, 256), 256, 0, stream>>>(pj_w, Wst, D_, D_);
    { GArgs g{}; g.A = A0; g.lda = D_; g.W = Wst; g.bias = pj_b;
      g.Mtiles = MT; g.Ntiles = D_/16; g.K = D_; g.H = H;
      k_gemm<EP_RES><<<(g.Mtiles/4)*g.Ntiles, 128, 0, stream>>>(g); }

    // MLP
    k_ln<<<MP, 256, 0, stream>>>(H, n2_w, n2_b, A0);
    k_wt<<<cdiv((long long)D_*MLPH_, 256), 256, 0, stream>>>(f1_w, Wst, D_, MLPH_);
    { GArgs g{}; g.A = A0; g.lda = D_; g.W = Wst; g.bias = f1_b;
      g.Mtiles = MT; g.Ntiles = MLPH_/16; g.K = D_; g.outbf = Amlp; g.ldo = MLPH_;
      k_gemm<EP_GELU><<<(g.Mtiles/4)*g.Ntiles, 128, 0, stream>>>(g); }
    k_wt<<<cdiv((long long)MLPH_*D_, 256), 256, 0, stream>>>(f2_w, Wst, MLPH_, D_);
    { GArgs g{}; g.A = Amlp; g.lda = MLPH_; g.W = Wst; g.bias = f2_b;
      g.Mtiles = MT; g.Ntiles = D_/16; g.K = MLPH_; g.H = H;
      k_gemm<EP_RES><<<(g.Mtiles/4)*g.Ntiles, 128, 0, stream>>>(g); }
  }

  k_head<<<1, 256, 0, stream>>>(H, norm_w, norm_b, head_w, head_b, (float*)d_out);
}
